// MMSingleStreamBlockReFuser_V3_15324443312843
// MI455X (gfx1250) — matrix operand order repaired and checked by reading, not compiled, not run on hardware
//
#include <hip/hip_runtime.h>
#include <math.h>

// ---------------- types ----------------
typedef __attribute__((ext_vector_type(16))) __bf16 v16bf;
typedef __attribute__((ext_vector_type(8)))  float  v8f;
typedef __attribute__((ext_vector_type(4)))  float  f4raw;

union FragBF { v16bf v; f4raw f[2]; };

// ---------------- problem constants ----------------
#define HID    3072
#define L_TOT  2304
#define IMG_N  2048
#define NH     24
#define DH     128
#define MLP_N  12288
#define REF_N  512
#define QKV_N  (3*HID)        // 9216
#define N1     (3*HID+MLP_N)  // 21504
#define N2CAT  (HID+MLP_N)    // 15360

#define EPI_LIN1  0
#define EPI_PLAIN 1
#define EPI_FINAL 2

// wave-local LDS ordering: same-wave DS ops retire in order; this only has to
// stop the compiler from reordering and drain DScnt before dependent reads.
#define WAVE_LDS_FENCE() asm volatile("s_wait_dscnt 0x0" ::: "memory")

// one-instruction packed f32x2 -> bf16x2 (low = lo, high = hi)
static __device__ __forceinline__ unsigned pack_bf16(float lo, float hi) {
  unsigned u;
  asm("v_cvt_pk_bf16_f32 %0, %1, %2" : "=v"(u) : "v"(lo), "v"(hi));
  return u;
}

// ---------------- small kernels ----------------

// mod = silu(vec) @ w_mod + b_mod   (K=3072, N=9216)
__global__ __launch_bounds__(256)
void k_silu_gemv(const float* __restrict__ vec, const float* __restrict__ wmod,
                 const float* __restrict__ bmod, float* __restrict__ mod) {
  __shared__ float sv[HID];
  for (int i = threadIdx.x; i < HID; i += 256) {
    float x = vec[i];
    sv[i] = x / (1.f + __expf(-x));
  }
  __syncthreads();
  int col = blockIdx.x * 256 + threadIdx.x;
  float acc = bmod[col];
  for (int k = 0; k < HID; ++k)
    acc = fmaf(sv[k], wmod[(size_t)k * QKV_N + col], acc);
  mod[col] = acc;
}

// x_mod = layernorm(x)*(1+scl)+shift  -> bf16 (packed pair stores)
__global__ __launch_bounds__(256)
void k_ln_mod(const float* __restrict__ x, const float* __restrict__ mod,
              __bf16* __restrict__ xmod) {
  int row = blockIdx.x;
  const float* xr = x + (size_t)row * HID;
  __shared__ float red[256];
  float s = 0.f;
  for (int c = threadIdx.x*2; c < HID; c += 512) s += xr[c] + xr[c+1];
  red[threadIdx.x] = s; __syncthreads();
  for (int o = 128; o > 0; o >>= 1) { if (threadIdx.x < o) red[threadIdx.x] += red[threadIdx.x+o]; __syncthreads(); }
  float mean = red[0] * (1.f/HID); __syncthreads();
  float v = 0.f;
  for (int c = threadIdx.x*2; c < HID; c += 512) {
    float d0 = xr[c]-mean, d1 = xr[c+1]-mean; v += d0*d0 + d1*d1;
  }
  red[threadIdx.x] = v; __syncthreads();
  for (int o = 128; o > 0; o >>= 1) { if (threadIdx.x < o) red[threadIdx.x] += red[threadIdx.x+o]; __syncthreads(); }
  float rstd = rsqrtf(red[0]*(1.f/HID) + 1e-6f);
  for (int c = threadIdx.x*2; c < HID; c += 512) {
    float v0 = (xr[c]  -mean)*rstd*(1.f+mod[HID+c])   + mod[c];
    float v1 = (xr[c+1]-mean)*rstd*(1.f+mod[HID+c+1]) + mod[c+1];
    *(unsigned*)&xmod[(size_t)row*HID + c] = pack_bf16(v0, v1);
  }
}

// ---------------- WMMA GEMM (double-buffered LDS) ----------------
// C[M,N] = A_bf16[M,K] @ B_f32[K,N] + bias, with epilogue modes.
__global__ __launch_bounds__(256)
void k_gemm(const __bf16* __restrict__ A, const float* __restrict__ Bw,
            const float* __restrict__ bias, int M, int N, int K, int mode,
            float* __restrict__ outF, __bf16* __restrict__ outBf,
            const float* __restrict__ xres, const float* __restrict__ gate,
            float* __restrict__ outFinal) {
  __shared__ __bf16 As[2][128][32];    // [buf][m][k]
  __shared__ __bf16 Bs[2][128][32];    // [buf][n][k] (transposed for contiguous B frags)
  int tid = threadIdx.x;
  int lane = tid & 31, wave = tid >> 5;
  int half = lane >> 4, ln = lane & 15;
  int mq = wave >> 1, nh = wave & 1;
  int M0 = blockIdx.y * 128, N0 = blockIdx.x * 128;

  v8f acc[2][4];
#pragma unroll
  for (int i = 0; i < 2; i++)
#pragma unroll
    for (int j = 0; j < 4; j++) acc[i][j] = v8f{0,0,0,0,0,0,0,0};

  int arow = tid >> 1, ahalf = tid & 1;   // A staging: 16 bf16 per thread
  int bkp  = tid >> 4, bnc  = tid & 15;   // B staging: 2(k) x 8(n) f32 per thread

  // staged-tile registers
  f4raw ra0, ra1, rx0, rx1, ry0, ry1;

  auto gload = [&](int k0) {
    const f4raw* src = (const f4raw*)(A + (size_t)(M0+arow)*K + k0 + ahalf*16);
    ra0 = src[0]; ra1 = src[1];
    const float* r0 = Bw + (size_t)(k0 + 2*bkp)*N + N0 + bnc*8;
    const float* r1 = r0 + N;
    rx0 = *(const f4raw*)r0; rx1 = *(const f4raw*)(r0+4);
    ry0 = *(const f4raw*)r1; ry1 = *(const f4raw*)(r1+4);
    if (k0 + 32 < K) {   // warm L2 one more tile ahead
      __builtin_prefetch(Bw + (size_t)(k0 + 32 + 2*bkp)*N + N0 + bnc*8, 0, 0);
      __builtin_prefetch(A + (size_t)(M0+arow)*K + k0 + 32, 0, 0);
    }
  };
  auto lstore = [&](int buf) {
    f4raw* dst = (f4raw*)&As[buf][arow][ahalf*16];
    dst[0] = ra0; dst[1] = ra1;
    float v0[8] = {rx0.x,rx0.y,rx0.z,rx0.w,rx1.x,rx1.y,rx1.z,rx1.w};
    float v1[8] = {ry0.x,ry0.y,ry0.z,ry0.w,ry1.x,ry1.y,ry1.z,ry1.w};
#pragma unroll
    for (int j = 0; j < 8; j++)   // one v_cvt_pk_bf16_f32 + one ds_store_b32 each
      *(unsigned*)&Bs[buf][bnc*8 + j][2*bkp] = pack_bf16(v0[j], v1[j]);
  };

  // prologue: stage tile 0 into buffer 0
  gload(0);
  lstore(0);
  int buf = 0;

  for (int k0 = 0; k0 < K; k0 += 32) {
    __syncthreads();                 // LDS[buf] visible to all waves
    bool more = (k0 + 32) < K;
    if (more) gload(k0 + 32);        // overlap next-tile loads with compute

    FragBF af[2], bfr[4];
#pragma unroll
    for (int mt = 0; mt < 2; mt++) {
      const __bf16* p = &As[buf][mq*32 + mt*16 + ln][half*16];
      af[mt].f[0] = *(const f4raw*)p;
      af[mt].f[1] = *(const f4raw*)(p + 8);
    }
#pragma unroll
    for (int nt = 0; nt < 4; nt++) {
      const __bf16* p = &Bs[buf][nh*64 + nt*16 + ln][half*16];
      bfr[nt].f[0] = *(const f4raw*)p;
      bfr[nt].f[1] = *(const f4raw*)(p + 8);
    }
#pragma unroll
    for (int mt = 0; mt < 2; mt++)
#pragma unroll
      for (int nt = 0; nt < 4; nt++)
        acc[mt][nt] = __builtin_amdgcn_wmma_f32_16x16x32_bf16(
            false, af[mt].v, false, bfr[nt].v, (short)0, acc[mt][nt], false, false);

    if (more) lstore(buf ^ 1);       // fill other buffer; guarded by next barrier
    buf ^= 1;
  }

  // epilogue
#pragma unroll
  for (int mt = 0; mt < 2; mt++)
#pragma unroll
    for (int nt = 0; nt < 4; nt++) {
      int grow0 = M0 + mq*32 + mt*16 + half*8;
      int gcol  = N0 + nh*64 + nt*16 + ln;
      float bb = bias[gcol];
#pragma unroll
      for (int r = 0; r < 8; r++) {
        float v = acc[mt][nt][r] + bb;
        int row = grow0 + r;
        if (mode == EPI_LIN1) {
          if (gcol < QKV_N) {
            outF[(size_t)row*QKV_N + gcol] = v;
          } else {
            float g = 0.5f*v*(1.f + tanhf(0.7978845608f*(v + 0.044715f*v*v*v)));
            outBf[(size_t)row*N2CAT + (HID + (gcol - QKV_N))] = (__bf16)g;
          }
        } else if (mode == EPI_PLAIN) {
          outF[(size_t)row*N + gcol] = v;
        } else {
          size_t idx = (size_t)row*HID + gcol;
          outFinal[idx] = xres[idx] + gate[gcol]*v;
        }
      }
    }
}

// ---------------- qkv post-processing: rmsnorm + rope (pairwise), relayout ----
// 64 threads/block: thread t owns the RoPE pair (d=2t, d=2t+1), so the rotation
// partner is in-thread and q/k stores are packed dwords.
__global__ __launch_bounds__(64)
void k_qkv(const float* __restrict__ qkv, const float* __restrict__ qnw,
           const float* __restrict__ knw, const float* __restrict__ cs,
           const float* __restrict__ sn,
           __bf16* __restrict__ qb, __bf16* __restrict__ kb, __bf16* __restrict__ vtb) {
  int l = blockIdx.x, h = blockIdx.y, t = threadIdx.x;
  int d0 = 2*t, d1 = 2*t + 1;
  const float* base = qkv + (size_t)l*QKV_N + h*DH;
  float q0 = base[d0],        q1 = base[d1];
  float k0 = base[HID+d0],    k1 = base[HID+d1];
  float v0 = base[2*HID+d0],  v1 = base[2*HID+d1];
  __shared__ float red[64];
  red[t] = q0*q0 + q1*q1; __syncthreads();
  for (int o = 32; o > 0; o >>= 1) { if (t < o) red[t] += red[t+o]; __syncthreads(); }
  float qr = rsqrtf(red[0]*(1.f/DH) + 1e-6f); __syncthreads();
  red[t] = k0*k0 + k1*k1; __syncthreads();
  for (int o = 32; o > 0; o >>= 1) { if (t < o) red[t] += red[t+o]; __syncthreads(); }
  float kr = rsqrtf(red[0]*(1.f/DH) + 1e-6f);
  float qn0 = q0*qr*qnw[d0], qn1 = q1*qr*qnw[d1];
  float kn0 = k0*kr*knw[d0], kn1 = k1*kr*knw[d1];
  float qo0 = qn0, qo1 = qn1, ko0 = kn0, ko1 = kn1;
  if (l < IMG_N) {
    float c = cs[(size_t)l*(DH/2) + t], s = sn[(size_t)l*(DH/2) + t];
    qo0 = qn0*c - qn1*s;  qo1 = qn0*s + qn1*c;
    ko0 = kn0*c - kn1*s;  ko1 = kn0*s + kn1*c;
  }
  *(unsigned*)&qb[((size_t)h*L_TOT + l)*DH + d0] = pack_bf16(qo0, qo1);
  *(unsigned*)&kb[((size_t)h*L_TOT + l)*DH + d0] = pack_bf16(ko0, ko1);
  vtb[((size_t)h*DH + d0)*L_TOT + l] = (__bf16)v0;   // d-major V
  vtb[((size_t)h*DH + d1)*L_TOT + l] = (__bf16)v1;
}

__global__ __launch_bounds__(64)
void k_refkv(const float* __restrict__ ktmp, const float* __restrict__ vtmp,
             const float* __restrict__ knw, const float* __restrict__ cs,
             const float* __restrict__ sn,
             __bf16* __restrict__ kb, __bf16* __restrict__ vtb) {
  int r = blockIdx.x, h = blockIdx.y, t = threadIdx.x;
  int d0 = 2*t, d1 = 2*t + 1;
  const float* kbase = ktmp + (size_t)r*HID + h*DH;
  const float* vbase = vtmp + (size_t)r*HID + h*DH;
  float k0 = kbase[d0], k1 = kbase[d1];
  float v0 = vbase[d0], v1 = vbase[d1];
  __shared__ float red[64];
  red[t] = k0*k0 + k1*k1; __syncthreads();
  for (int o = 32; o > 0; o >>= 1) { if (t < o) red[t] += red[t+o]; __syncthreads(); }
  float kr = rsqrtf(red[0]*(1.f/DH) + 1e-6f);
  float kn0 = k0*kr*knw[d0], kn1 = k1*kr*knw[d1];
  float c = cs[(size_t)r*(DH/2) + t], s = sn[(size_t)r*(DH/2) + t];
  float ko0 = kn0*c - kn1*s, ko1 = kn0*s + kn1*c;
  *(unsigned*)&kb[((size_t)h*REF_N + r)*DH + d0] = pack_bf16(ko0, ko1);
  vtb[((size_t)h*DH + d0)*REF_N + r] = (__bf16)v0;
  vtb[((size_t)h*DH + d1)*REF_N + r] = (__bf16)v1;
}

// ---------------- flash attention (WMMA) ----------------
// Q,K: [H][L][128] bf16; Vt: [H][128][Lk] bf16; Out: [Lq][H*128] f32
// Each wave owns 16 query rows and a private LDS transpose buffer; no block
// barriers needed — same-wave DS ordering + s_wait_dscnt suffices.
__global__ __launch_bounds__(256)
void k_attn(const __bf16* __restrict__ Q, const __bf16* __restrict__ Km,
            const __bf16* __restrict__ Vt, float* __restrict__ Out,
            int Lk, int accumulate) {
  int h = blockIdx.y;
  int tid = threadIdx.x, wave = tid >> 5, lane = tid & 31;
  int half = lane >> 4, ln = lane & 15;
  int qrow0 = blockIdx.x*128 + wave*16;
  const float scale = 0.08838834764831845f; // 1/sqrt(128)

  __shared__ __bf16 psh[8][16][32];         // per-wave P transpose buffer

  FragBF qf[4];
#pragma unroll
  for (int c = 0; c < 4; c++) {
    const __bf16* p = Q + ((size_t)h*L_TOT + qrow0 + ln)*DH + c*32 + half*16;
    qf[c].f[0] = *(const f4raw*)p;
    qf[c].f[1] = *(const f4raw*)(p + 8);
  }

  v8f o[8];
#pragma unroll
  for (int t = 0; t < 8; t++) o[t] = v8f{0,0,0,0,0,0,0,0};
  float mrow[8], lrow[8];
#pragma unroll
  for (int r = 0; r < 8; r++) { mrow[r] = -1e30f; lrow[r] = 0.f; }

  for (int kbase = 0; kbase < Lk; kbase += 32) {
    v8f s0 = v8f{0,0,0,0,0,0,0,0}, s1 = v8f{0,0,0,0,0,0,0,0};
#pragma unroll
    for (int c = 0; c < 4; c++) {
      FragBF kf0, kf1;
      const __bf16* p0 = Km + ((size_t)h*Lk + kbase + ln)*DH + c*32 + half*16;
      kf0.f[0] = *(const f4raw*)p0;       kf0.f[1] = *(const f4raw*)(p0 + 8);
      const __bf16* p1 = p0 + 16*DH;
      kf1.f[0] = *(const f4raw*)p1;       kf1.f[1] = *(const f4raw*)(p1 + 8);
      s0 = __builtin_amdgcn_wmma_f32_16x16x32_bf16(false, qf[c].v, false, kf0.v, (short)0, s0, false, false);
      s1 = __builtin_amdgcn_wmma_f32_16x16x32_bf16(false, qf[c].v, false, kf1.v, (short)0, s1, false, false);
    }
    // online softmax, per C-row r (qrow = r + 8*half), stats shared across 16-lane half
#pragma unroll
    for (int r = 0; r < 8; r++) {
      float a0 = s0[r]*scale, a1 = s1[r]*scale;
      float mx = fmaxf(a0, a1);
#pragma unroll
      for (int msk = 1; msk < 16; msk <<= 1) mx = fmaxf(mx, __shfl_xor(mx, msk, 32));
      float mnew = fmaxf(mrow[r], mx);
      float corr = __expf(mrow[r] - mnew);
      float p0 = __expf(a0 - mnew), p1 = __expf(a1 - mnew);
      float ps = p0 + p1;
#pragma unroll
      for (int msk = 1; msk < 16; msk <<= 1) ps += __shfl_xor(ps, msk, 32);
      lrow[r] = lrow[r]*corr + ps;
      mrow[r] = mnew;
#pragma unroll
      for (int t = 0; t < 8; t++) o[t][r] *= corr;
      int qr = r + half*8;
      psh[wave][qr][ln]      = (__bf16)p0;
      psh[wave][qr][16 + ln] = (__bf16)p1;
    }
    WAVE_LDS_FENCE();                  // wave-local: stores drained before reads
    FragBF pf;
    {
      const __bf16* p = &psh[wave][ln][half*16];
      pf.f[0] = *(const f4raw*)p;
      pf.f[1] = *(const f4raw*)(p + 8);
    }
    WAVE_LDS_FENCE();                  // reads drained before next-iter stores
#pragma unroll
    for (int t = 0; t < 8; t++) {
      FragBF vf;
      const __bf16* p = Vt + ((size_t)h*DH + t*16 + ln)*Lk + kbase + half*16;
      vf.f[0] = *(const f4raw*)p;
      vf.f[1] = *(const f4raw*)(p + 8);
      o[t] = __builtin_amdgcn_wmma_f32_16x16x32_bf16(false, pf.v, false, vf.v, (short)0, o[t], false, false);
    }
  }

#pragma unroll
  for (int t = 0; t < 8; t++)
#pragma unroll
    for (int r = 0; r < 8; r++) {
      int row = qrow0 + r + half*8;
      int d   = t*16 + ln;
      float val = o[t][r] / lrow[r];
      size_t idx = (size_t)row*HID + h*DH + d;
      if (accumulate) Out[idx] += val; else Out[idx] = val;
    }
}

// ---------------- converts (packed pair per thread) ----------------
__global__ __launch_bounds__(256)
void k_f2bf(const float* __restrict__ src, __bf16* __restrict__ dst, int n) {
  int i = (blockIdx.x*256 + threadIdx.x)*2;
  if (i < n)
    *(unsigned*)&dst[i] = pack_bf16(src[i], src[i+1]);
}

__global__ __launch_bounds__(256)
void k_attn2A2(const float* __restrict__ attn, __bf16* __restrict__ A2) {
  int i = (blockIdx.x*256 + threadIdx.x)*2;   // n = 2304*3072, HID even
  int row = i / HID, col = i - row*HID;
  *(unsigned*)&A2[(size_t)row*N2CAT + col] = pack_bf16(attn[i], attn[i+1]);
}

// ---------------- launch ----------------
extern "C" void kernel_launch(void* const* d_in, const int* in_sizes, int n_in,
                              void* d_out, int out_size, void* d_ws, size_t ws_size,
                              hipStream_t stream) {
  (void)in_sizes; (void)n_in; (void)out_size; (void)ws_size;
  const float* x    = (const float*)d_in[0];
  const float* vec  = (const float*)d_in[1];
  const float* cosp = (const float*)d_in[2];
  const float* sinp = (const float*)d_in[3];
  const float* refl = (const float*)d_in[4];
  const float* rcos = (const float*)d_in[5];
  const float* rsin = (const float*)d_in[6];
  const float* wmod = (const float*)d_in[7];
  const float* bmod = (const float*)d_in[8];
  const float* w1   = (const float*)d_in[9];
  const float* b1   = (const float*)d_in[10];
  const float* w2   = (const float*)d_in[11];
  const float* b2   = (const float*)d_in[12];
  const float* qnw  = (const float*)d_in[13];
  const float* knw  = (const float*)d_in[14];
  const float* wkip = (const float*)d_in[15];
  const float* bkip = (const float*)d_in[16];
  const float* wvip = (const float*)d_in[17];
  const float* bvip = (const float*)d_in[18];
  const float* rknw = (const float*)d_in[19];
  float* out = (float*)d_out;

  char* w = (char*)d_ws;
  size_t off = 0;
  auto take = [&](size_t bytes) -> void* {
    void* p = w + off; off += (bytes + 255) & ~(size_t)255; return p;
  };
  float*  mod   = (float*) take((size_t)QKV_N*4);
  __bf16* xmod  = (__bf16*)take((size_t)L_TOT*HID*2);
  float*  qkvf  = (float*) take((size_t)L_TOT*QKV_N*4);
  __bf16* qb    = (__bf16*)take((size_t)NH*L_TOT*DH*2);
  __bf16* kb    = (__bf16*)take((size_t)NH*L_TOT*DH*2);
  __bf16* vtb   = (__bf16*)take((size_t)NH*DH*L_TOT*2);
  __bf16* refA  = (__bf16*)take((size_t)REF_N*HID*2);
  float*  rktmp = (float*) take((size_t)REF_N*HID*4);
  float*  rvtmp = (float*) take((size_t)REF_N*HID*4);
  __bf16* rkb   = (__bf16*)take((size_t)NH*REF_N*DH*2);
  __bf16* rvtb  = (__bf16*)take((size_t)NH*DH*REF_N*2);
  float*  attnf = (float*) take((size_t)L_TOT*HID*4);
  __bf16* A2    = (__bf16*)take((size_t)L_TOT*N2CAT*2);

  // 1) modulation vector
  k_silu_gemv<<<QKV_N/256, 256, 0, stream>>>(vec, wmod, bmod, mod);
  // 2) layernorm + modulate -> bf16
  k_ln_mod<<<L_TOT, 256, 0, stream>>>(x, mod, xmod);
  // 3) lin1: qkv -> f32, gelu(mlp) -> bf16 straight into A2 concat buffer
  k_gemm<<<dim3(N1/128, L_TOT/128), 256, 0, stream>>>(
      xmod, w1, b1, L_TOT, N1, HID, EPI_LIN1, qkvf, A2, nullptr, nullptr, nullptr);
  // 4) rmsnorm + rope + relayout q/k/v
  k_qkv<<<dim3(L_TOT, NH), 64, 0, stream>>>(qkvf, qnw, knw, cosp, sinp, qb, kb, vtb);
  // 5) ref_latent -> bf16, project k/v
  k_f2bf<<<(REF_N*HID)/512, 256, 0, stream>>>(refl, refA, REF_N*HID);
  k_gemm<<<dim3(HID/128, REF_N/128), 256, 0, stream>>>(
      refA, wkip, bkip, REF_N, HID, HID, EPI_PLAIN, rktmp, nullptr, nullptr, nullptr, nullptr);
  k_gemm<<<dim3(HID/128, REF_N/128), 256, 0, stream>>>(
      refA, wvip, bvip, REF_N, HID, HID, EPI_PLAIN, rvtmp, nullptr, nullptr, nullptr, nullptr);
  k_refkv<<<dim3(REF_N, NH), 64, 0, stream>>>(rktmp, rvtmp, rknw, rcos, rsin, rkb, rvtb);
  // 6) self-attention + ip-attention (accumulated)
  k_attn<<<dim3(L_TOT/128, NH), 256, 0, stream>>>(qb, kb,  vtb,  attnf, L_TOT, 0);
  k_attn<<<dim3(L_TOT/128, NH), 256, 0, stream>>>(qb, rkb, rvtb, attnf, REF_N, 1);
  // 7) attn -> bf16 into A2[:, :3072]
  k_attn2A2<<<(L_TOT*HID)/512, 256, 0, stream>>>(attnf, A2);
  // 8) out projection + gate + residual
  k_gemm<<<dim3(HID/128, L_TOT/128), 256, 0, stream>>>(
      A2, w2, b2, L_TOT, HID, N2CAT, EPI_FINAL, nullptr, nullptr, x, mod + 2*HID, out);
}
